// DeCooper_39152921870885
// MI455X (gfx1250) — matrix operand order repaired and checked by reading, not compiled, not run on hardware
//
#include <hip/hip_runtime.h>
#include <hip/hip_bf16.h>

typedef __attribute__((ext_vector_type(16))) _Float16 v16h;
typedef __attribute__((ext_vector_type(8)))  _Float16 v8h;
typedef __attribute__((ext_vector_type(8)))  float    v8f;

#define DIMC   256
#define NLV    3
#define NHEAD  8
#define NPT    4
#define HDIM   32
#define BATCH  16
#define LTOT   2772           // 2112 + 528 + 132
#define MROWS  (BATCH * LTOT) // 44352
#define NCOMB  288            // 192 off + 96 attn

// ---------------- level geometry helpers ----------------
static __device__ __forceinline__ void decode_pos(int pos, int& lvl, int& p,
                                                  int& H, int& W) {
  if (pos < 2112)      { lvl = 0; p = pos;        H = 24; W = 88; }
  else if (pos < 2640) { lvl = 1; p = pos - 2112; H = 12; W = 44; }
  else                 { lvl = 2; p = pos - 2640; H = 6;  W = 22; }
}

// ---------------- WMMA fragment loaders ----------------
// A: 16x32 f16 tile, row m in lane (lane&15), K split per ISA layout:
// lanes 0-15: e0..7 -> K k0+0..7,  e8..15 -> K k0+16..23
// lanes16-31: e0..7 -> K k0+8..15, e8..15 -> K k0+24..31
static __device__ __forceinline__ v16h load_fragA(const _Float16* base, int row0,
                                                  int k0, int lane) {
  int m  = lane & 15;
  int hi = (lane >> 4) << 3;
  const _Float16* p = base + (size_t)(row0 + m) * DIMC + k0 + hi;
  v8h lo = *(const v8h*)(p);
  v8h up = *(const v8h*)(p + 16);
  return __builtin_shufflevector(lo, up, 0,1,2,3,4,5,6,7,8,9,10,11,12,13,14,15);
}

// B: 32x16 f16 tile, column n = lane&15; same K split as A:
// lanes 0-15: K k0+0..7, k0+16..23; lanes 16-31: K k0+8..15, k0+24..31.
// Weight stored row-major W[n][k]  (B = W^T so B[K][n] = W[n][K]).
static __device__ __forceinline__ v16h load_fragB(const _Float16* Wt, int n0,
                                                  int k0, int lane) {
  int n  = lane & 15;
  int kh = (lane >> 4) << 3;
  const _Float16* p = Wt + (size_t)(n0 + n) * DIMC + k0 + kh;
  v8h lo = *(const v8h*)(p);
  v8h up = *(const v8h*)(p + 16);
  return __builtin_shufflevector(lo, up, 0,1,2,3,4,5,6,7,8,9,10,11,12,13,14,15);
}

static __device__ __forceinline__ v8f wmma_f16(v16h a, v16h b, v8f c) {
  return __builtin_amdgcn_wmma_f32_16x16x32_f16(false, a, false, b,
                                                (short)0, c, false, false);
}

// 32x32 output tile per wave: 2x2 WMMA tiles, K = 256 unrolled.
static __device__ __forceinline__ void gemm_tile32(const _Float16* A,
                                                   const _Float16* Bw,
                                                   int m0, int n0, int lane,
                                                   v8f c[4]) {
  v8f z{};
  c[0] = z; c[1] = z; c[2] = z; c[3] = z;
#pragma unroll
  for (int k = 0; k < DIMC; k += 32) {
    v16h a0 = load_fragA(A, m0,      k, lane);
    v16h a1 = load_fragA(A, m0 + 16, k, lane);
    v16h b0 = load_fragB(Bw, n0,      k, lane);
    v16h b1 = load_fragB(Bw, n0 + 16, k, lane);
    c[0] = wmma_f16(a0, b0, c[0]);
    c[1] = wmma_f16(a0, b1, c[1]);
    c[2] = wmma_f16(a1, b0, c[2]);
    c[3] = wmma_f16(a1, b1, c[3]);
  }
}

// ---------------- kernel 1: weights -> f16 ----------------
__global__ void prep_weights(const float* __restrict__ Wval,
                             const float* __restrict__ Woff,
                             const float* __restrict__ Wattn,
                             const float* __restrict__ Wout,
                             _Float16* __restrict__ wval16,
                             _Float16* __restrict__ wcomb16,
                             _Float16* __restrict__ wout16) {
  int idx = blockIdx.x * blockDim.x + threadIdx.x;
  const int NV = DIMC * DIMC;        // 65536
  const int NC = NCOMB * DIMC;       // 73728
  if (idx < NV) { wval16[idx] = (_Float16)Wval[idx]; return; }
  idx -= NV;
  if (idx < NC) {
    int n = idx >> 8, k = idx & 255;
    float v = (n < 192) ? Woff[n * DIMC + k] : Wattn[(n - 192) * DIMC + k];
    wcomb16[idx] = (_Float16)v;
    return;
  }
  idx -= NC;
  if (idx < NV) wout16[idx] = (_Float16)Wout[idx];
}

// ---------------- kernel 2: transpose + sine pos embed ----------------
__global__ void prep_act(const float* __restrict__ q0, const float* __restrict__ q1,
                         const float* __restrict__ q2, const float* __restrict__ f0,
                         const float* __restrict__ f1, const float* __restrict__ f2,
                         _Float16* __restrict__ qloc16, _Float16* __restrict__ inp16) {
  int idx = blockIdx.x * blockDim.x + threadIdx.x;  // MROWS*256 threads
  int c   = idx & 255;
  int row = idx >> 8;
  if (row >= MROWS) return;
  int b   = row / LTOT;
  int pos = row - b * LTOT;
  int lvl, p, H, W;
  decode_pos(pos, lvl, p, H, W);
  int h = p / W, w = p - h * W;
  const float* q = (lvl == 0) ? q0 : (lvl == 1) ? q1 : q2;
  const float* f = (lvl == 0) ? f0 : (lvl == 1) ? f1 : f2;
  size_t off = ((size_t)(b * DIMC + c) * H + h) * W + w;
  float fv = f[off];
  float qv = q[off];
  // sine positional embedding: ch 0..127 -> y part, 128..255 -> x part
  int   j     = (c < 128) ? c : (c - 128);
  float coord = (c < 128) ? (float)(h + 1) : (float)(w + 1);
  int   pr    = j >> 1;
  float denom = __powf(10000.0f, (float)pr * (1.0f / 64.0f));
  float arg   = coord / denom;
  float pe    = (j & 1) ? __cosf(arg) : __sinf(arg);
  qloc16[idx] = (_Float16)(qv + pe);
  inp16[idx]  = (_Float16)fv;
}

// ---------------- kernel 3: value GEMM (N=256, f16 out) ----------------
__global__ void gemm_value(const _Float16* __restrict__ A,
                           const _Float16* __restrict__ Bw,
                           const float* __restrict__ bias,
                           _Float16* __restrict__ out) {
  int gw   = (blockIdx.x * blockDim.x + threadIdx.x) >> 5;
  int lane = threadIdx.x & 31;
  const int NT = DIMC / 32;  // 8
  int nt = gw % NT, mt = gw / NT;
  if (mt >= MROWS / 32) return;
  int m0 = mt * 32, n0 = nt * 32;
  v8f c[4];
  gemm_tile32(A, Bw, m0, n0, lane, c);
  int nl = lane & 15;
  int mh = (lane >> 4) * 8;
  float b0 = bias[n0 + nl], b1 = bias[n0 + 16 + nl];
#pragma unroll
  for (int r = 0; r < 8; ++r) {
    int ma = m0 + mh + r;
    int mb = ma + 16;
    out[(size_t)ma * DIMC + n0 + nl]      = (_Float16)(c[0][r] + b0);
    out[(size_t)ma * DIMC + n0 + 16 + nl] = (_Float16)(c[1][r] + b1);
    out[(size_t)mb * DIMC + n0 + nl]      = (_Float16)(c[2][r] + b0);
    out[(size_t)mb * DIMC + n0 + 16 + nl] = (_Float16)(c[3][r] + b1);
  }
}

// ---------------- kernel 4: off+attn GEMM (N=288, f32 out) ----------------
__global__ void gemm_offattn(const _Float16* __restrict__ A,
                             const _Float16* __restrict__ Bw,
                             const float* __restrict__ boff,
                             const float* __restrict__ battn,
                             float* __restrict__ out) {
  int gw   = (blockIdx.x * blockDim.x + threadIdx.x) >> 5;
  int lane = threadIdx.x & 31;
  const int NT = NCOMB / 32;  // 9
  if (gw >= (MROWS / 32) * NT) return;
  int nt = gw % NT, mt = gw / NT;
  int m0 = mt * 32, n0 = nt * 32;
  v8f c[4];
  gemm_tile32(A, Bw, m0, n0, lane, c);
  int nl = lane & 15;
  int mh = (lane >> 4) * 8;
  int na = n0 + nl, nb = n0 + 16 + nl;
  float ba = (na < 192) ? boff[na] : battn[na - 192];
  float bb = (nb < 192) ? boff[nb] : battn[nb - 192];
#pragma unroll
  for (int r = 0; r < 8; ++r) {
    int ma = m0 + mh + r;
    int mb = ma + 16;
    out[(size_t)ma * NCOMB + na] = c[0][r] + ba;
    out[(size_t)ma * NCOMB + nb] = c[1][r] + bb;
    out[(size_t)mb * NCOMB + na] = c[2][r] + ba;
    out[(size_t)mb * NCOMB + nb] = c[3][r] + bb;
  }
}

// ---------------- kernel 5: softmax over 12 (level,point) per head ----------
__global__ void softmax12(const float* __restrict__ offattn,
                          float* __restrict__ aw) {
  int idx = blockIdx.x * blockDim.x + threadIdx.x;  // MROWS*NHEAD
  if (idx >= MROWS * NHEAD) return;
  int row = idx >> 3, h = idx & 7;
  const float* src = offattn + (size_t)row * NCOMB + 192 + h * 12;
  float v[12];
  float mx = -1e30f;
#pragma unroll
  for (int i = 0; i < 12; ++i) { v[i] = src[i]; mx = fmaxf(mx, v[i]); }
  float s = 0.f;
#pragma unroll
  for (int i = 0; i < 12; ++i) { v[i] = __expf(v[i] - mx); s += v[i]; }
  float inv = 1.0f / s;
  float* dst = aw + (size_t)row * 96 + h * 12;
#pragma unroll
  for (int i = 0; i < 12; ++i) dst[i] = v[i] * inv;
}

// ---------------- kernel 6: deformable bilinear sampling ----------------
// wave = (b, query, head); lane = head-dim channel (HDIM == 32 == wave32)
__global__ void msda_sample(const _Float16* __restrict__ value,
                            const float* __restrict__ offattn,
                            const float* __restrict__ aw,
                            _Float16* __restrict__ out) {
  int gw   = (blockIdx.x * blockDim.x + threadIdx.x) >> 5;
  int lane = threadIdx.x & 31;
  if (gw >= MROWS * NHEAD) return;
  int h   = gw & 7;
  int row = gw >> 3;
  int b   = row / LTOT;
  int pos = row - b * LTOT;
  int qlvl, qp, qH, qW;
  decode_pos(pos, qlvl, qp, qH, qW);
  int hq = qp / qW, wq = qp - hq * qW;
  float refx = ((float)wq + 0.5f) / (float)qW;
  float refy = ((float)hq + 0.5f) / (float)qH;
  const float* offp = offattn + (size_t)row * NCOMB + h * 24;
  const float* awp  = aw + (size_t)row * 96 + h * 12;
  const int Hs[NLV] = {24, 12, 6};
  const int Ws[NLV] = {88, 44, 22};
  const int Ss[NLV] = {0, 2112, 2640};
  float acc = 0.f;
#pragma unroll
  for (int lvl = 0; lvl < NLV; ++lvl) {
    int H = Hs[lvl], W = Ws[lvl];
    size_t rowbase = ((size_t)(b * LTOT + Ss[lvl])) * DIMC + h * HDIM + lane;
#pragma unroll
    for (int pt = 0; pt < NPT; ++pt) {
      float ox = offp[lvl * 8 + pt * 2 + 0];
      float oy = offp[lvl * 8 + pt * 2 + 1];
      float a  = awp[lvl * 4 + pt];
      float x = (refx + ox / (float)W) * (float)W - 0.5f;
      float y = (refy + oy / (float)H) * (float)H - 0.5f;
      float x0f = floorf(x), y0f = floorf(y);
      int   x0 = (int)x0f,  y0 = (int)y0f;
      float dx = x - x0f,   dy = y - y0f;
      float s = 0.f;
#pragma unroll
      for (int cy = 0; cy < 2; ++cy) {
#pragma unroll
        for (int cx = 0; cx < 2; ++cx) {
          int xi = x0 + cx, yi = y0 + cy;
          float wgt = (cx ? dx : 1.f - dx) * (cy ? dy : 1.f - dy);
          if (xi >= 0 && xi < W && yi >= 0 && yi < H) {  // wave-uniform branch
            float v = (float)value[rowbase + (size_t)(yi * W + xi) * DIMC];
            s += wgt * v;
          }
        }
      }
      acc += a * s;
    }
  }
  out[(size_t)row * DIMC + h * HDIM + lane] = (_Float16)acc;
}

// ---------------- kernel 7: output GEMM + scatter to (B,C,H,W) ------------
static __device__ __forceinline__ void store_out(float* __restrict__ out,
                                                 int m, int n, float val) {
  int b   = m / LTOT;
  int pos = m - b * LTOT;
  int lvl, p, H, W;
  decode_pos(pos, lvl, p, H, W);
  int hh = p / W, ww = p - hh * W;
  size_t base = (lvl == 0) ? 0 : (lvl == 1) ? (size_t)8650752 : (size_t)10813440;
  out[base + ((size_t)(b * DIMC + n) * H + hh) * W + ww] = val;
}

__global__ void gemm_out(const _Float16* __restrict__ A,
                         const _Float16* __restrict__ Bw,
                         const float* __restrict__ bias,
                         float* __restrict__ out) {
  int gw   = (blockIdx.x * blockDim.x + threadIdx.x) >> 5;
  int lane = threadIdx.x & 31;
  const int NT = DIMC / 32;  // 8
  int nt = gw % NT, mt = gw / NT;
  if (mt >= MROWS / 32) return;
  int m0 = mt * 32, n0 = nt * 32;
  v8f c[4];
  gemm_tile32(A, Bw, m0, n0, lane, c);
  int nl = lane & 15;
  int mh = (lane >> 4) * 8;
  int na = n0 + nl, nb = n0 + 16 + nl;
  float ba = bias[na], bb = bias[nb];
#pragma unroll
  for (int r = 0; r < 8; ++r) {
    int ma = m0 + mh + r;
    int mb = ma + 16;
    store_out(out, ma, na, c[0][r] + ba);
    store_out(out, ma, nb, c[1][r] + bb);
    store_out(out, mb, na, c[2][r] + ba);
    store_out(out, mb, nb, c[3][r] + bb);
  }
}

// ---------------- launcher ----------------
extern "C" void kernel_launch(void* const* d_in, const int* in_sizes, int n_in,
                              void* d_out, int out_size, void* d_ws, size_t ws_size,
                              hipStream_t stream) {
  const float* q0    = (const float*)d_in[0];
  const float* q1    = (const float*)d_in[1];
  const float* q2    = (const float*)d_in[2];
  const float* f0    = (const float*)d_in[3];
  const float* f1    = (const float*)d_in[4];
  const float* f2    = (const float*)d_in[5];
  const float* Woff  = (const float*)d_in[6];
  const float* boff  = (const float*)d_in[7];
  const float* Wattn = (const float*)d_in[8];
  const float* battn = (const float*)d_in[9];
  const float* Wval  = (const float*)d_in[10];
  const float* bval  = (const float*)d_in[11];
  const float* Wout  = (const float*)d_in[12];
  const float* bout  = (const float*)d_in[13];
  float* out = (float*)d_out;

  char* ws = (char*)d_ws;
  const size_t sz_act     = (size_t)MROWS * DIMC * sizeof(_Float16);  // 22,708,224
  const size_t sz_offattn = (size_t)MROWS * NCOMB * sizeof(float);    // 51,093,504
  const size_t sz_aw      = (size_t)MROWS * 96 * sizeof(float);       // 17,031,168
  _Float16* qloc16    = (_Float16*)(ws);
  _Float16* inp16     = (_Float16*)(ws + sz_act);
  _Float16* value16   = (_Float16*)(ws + 2 * sz_act);
  _Float16* attnout16 = (_Float16*)(ws + 3 * sz_act);
  float*    offattn   = (float*)(ws + 4 * sz_act);
  float*    aw        = (float*)(ws + 4 * sz_act + sz_offattn);
  char*     wbase     = ws + 4 * sz_act + sz_offattn + sz_aw;
  _Float16* wval16    = (_Float16*)(wbase);
  _Float16* wcomb16   = (_Float16*)(wbase + (size_t)DIMC * DIMC * 2);
  _Float16* wout16    = (_Float16*)(wbase + (size_t)DIMC * DIMC * 2 +
                                    (size_t)NCOMB * DIMC * 2);

  // 1) weights -> f16              (204800 elements)
  prep_weights<<<800, 256, 0, stream>>>(Wval, Woff, Wattn, Wout,
                                        wval16, wcomb16, wout16);
  // 2) transpose + positional embed (MROWS*256 threads)
  prep_act<<<MROWS, 256, 0, stream>>>(q0, q1, q2, f0, f1, f2, qloc16, inp16);
  // 3) value GEMM: 1386 * 8 tiles, 8 waves/block
  gemm_value<<<(MROWS / 32) * (DIMC / 32) / 8, 256, 0, stream>>>(
      inp16, wval16, bval, value16);
  // 4) off+attn GEMM: 1386 * 9 tiles
  {
    int tiles = (MROWS / 32) * (NCOMB / 32);
    gemm_offattn<<<(tiles + 7) / 8, 256, 0, stream>>>(qloc16, wcomb16, boff,
                                                      battn, offattn);
  }
  // 5) softmax: MROWS*NHEAD threads
  softmax12<<<(MROWS * NHEAD) / 256, 256, 0, stream>>>(offattn, aw);
  // 6) sampling: MROWS*NHEAD waves, 8 waves/block
  msda_sample<<<(MROWS * NHEAD) / 8, 256, 0, stream>>>(value16, offattn, aw,
                                                       attnout16);
  // 7) output GEMM + scatter
  gemm_out<<<(MROWS / 32) * (DIMC / 32) / 8, 256, 0, stream>>>(
      attnout16, wout16, bout, out);
}
